// DCAELiteMLA_75213467287693
// MI455X (gfx1250) — matrix operand order repaired and checked by reading, not compiled, not run on hardware
//
#include <hip/hip_runtime.h>
#include <hip/hip_bf16.h>

typedef __attribute__((ext_vector_type(16))) __bf16 v16bf;
typedef __attribute__((ext_vector_type(8)))  __bf16 v8bf;
typedef __attribute__((ext_vector_type(2)))  __bf16 v2bf;
typedef __attribute__((ext_vector_type(8)))  float  v8f;

#define TILE_M 128
#define TILE_N 128
#define TILE_K 32
#define LDSS   40   // bf16 elements per LDS row (80B: 16B aligned, bank-spread)

// Native f32->bf16 (RNE). Prefer packed HW convert; fall back to fptrunc.
__device__ __forceinline__ v2bf f2bf2(float x, float y) {
#if __has_builtin(__builtin_amdgcn_cvt_pk_bf16_f32)
    return __builtin_amdgcn_cvt_pk_bf16_f32(x, y);
#else
    v2bf r; r[0] = (__bf16)x; r[1] = (__bf16)y; return r;
#endif
}
__device__ __forceinline__ __bf16 f2bf(float f) { return (__bf16)f; }

__device__ __forceinline__ v16bf load_frag(const __bf16* lo, const __bf16* hi) {
    v8bf a = *(const v8bf*)lo;
    v8bf b = *(const v8bf*)hi;
    v16bf r;
#pragma unroll
    for (int t = 0; t < 8; ++t) { r[t] = a[t]; r[t + 8] = b[t]; }
    return r;
}

// Stage one K-tile's worth of global data into registers (A: 128x32, B: 32x128).
__device__ __forceinline__ void load_tiles(const float* __restrict__ W,
                                           const float* __restrict__ Xb,
                                           int K, int m_blk, int n_blk, int kt,
                                           int tid, float4 a[4], float4 b[4]) {
#pragma unroll
    for (int u = 0; u < 4; ++u) {           // A: 1024 float4, 4 per thread
        const int i  = tid + u * 256;
        const int r  = i >> 3;              // 8 float4 per 32-wide row
        const int cq = (i & 7) << 2;
        a[u] = *(const float4*)(W + (long)(m_blk + r) * K + kt + cq);
    }
#pragma unroll
    for (int u = 0; u < 4; ++u) {           // B: 1024 float4, 4 per thread
        const int i  = tid + u * 256;
        const int kr = i >> 5;              // 32 float4 per 128-wide row
        const int nc = (i & 31) << 2;
        b[u] = *(const float4*)(Xb + (long)(kt + kr) * 4096 + n_blk + nc);
    }
}

// Convert fp32 registers -> bf16 LDS tiles (B stored transposed: Bls[n][k]).
__device__ __forceinline__ void store_tiles(__bf16* __restrict__ Als,
                                            __bf16* __restrict__ Bls,
                                            int tid, const float4 a[4], const float4 b[4]) {
#pragma unroll
    for (int u = 0; u < 4; ++u) {
        const int i  = tid + u * 256;
        const int r  = i >> 3;
        const int cq = (i & 7) << 2;
        __bf16* d = &Als[r * LDSS + cq];
        *(v2bf*)(d)     = f2bf2(a[u].x, a[u].y);   // packed cvt + b32 store
        *(v2bf*)(d + 2) = f2bf2(a[u].z, a[u].w);
    }
#pragma unroll
    for (int u = 0; u < 4; ++u) {
        const int i  = tid + u * 256;
        const int kr = i >> 5;
        const int nc = (i & 31) << 2;
        Bls[(nc + 0) * LDSS + kr] = f2bf(b[u].x);  // 4 distinct rows: scalar stores
        Bls[(nc + 1) * LDSS + kr] = f2bf(b[u].y);
        Bls[(nc + 2) * LDSS + kr] = f2bf(b[u].z);
        Bls[(nc + 3) * LDSS + kr] = f2bf(b[u].w);
    }
}

// C[b, chOff+m, n] = sum_k W[m,k] * X[b,k,n]  (+ optional residual R[b,m,n])
// Double-buffered LDS, software-pipelined: globals for tile t+1 issue before
// the WMMAs of tile t; convert+ds_store land behind the matrix ops.
__global__ __launch_bounds__(256) void gemm_wmma(
    const float* __restrict__ W,     // [M][K] row-major
    const float* __restrict__ X,     // [B][K][4096]
    float*       __restrict__ C,     // [B][cChan][4096]
    const float* __restrict__ R,     // [B][M][4096] or nullptr
    int M, int K, int cChan, int chOff)
{
    __shared__ __bf16 AlsBuf[2][TILE_M * LDSS];
    __shared__ __bf16 BlsBuf[2][TILE_N * LDSS];

    const int tid  = threadIdx.x;
    const int lane = tid & 31;
    const int wave = tid >> 5;
    const int b     = blockIdx.z;
    const int m_blk = blockIdx.y * TILE_M;
    const int n_blk = blockIdx.x * TILE_N;

    const int mw = (wave & 3) * 32;   // wave M offset (4 waves over 128)
    const int nw = (wave >> 2) * 64;  // wave N offset (2 waves over 128)

    const float* Xb = X + (long)b * K * 4096;

    v8f acc[2][4] = {};
    float4 ar[4], br[4];

    // prologue: stage tile 0
    load_tiles(W, Xb, K, m_blk, n_blk, 0, tid, ar, br);
    store_tiles(AlsBuf[0], BlsBuf[0], tid, ar, br);

    const int T = K / TILE_K;
    for (int t = 0; t < T; ++t) {
        __syncthreads();   // tile t visible; prior readers of buffer (t+1)&1 done

        if (t + 1 < T)     // issue next tile's global loads early (latency shadow)
            load_tiles(W, Xb, K, m_blk, n_blk, (t + 1) * TILE_K, tid, ar, br);

        const __bf16* Als = AlsBuf[t & 1];
        const __bf16* Bls = BlsBuf[t & 1];

        // A frag (16x32 bf16): lanes 0-15 -> row=lane, K 0..7 & 16..23;
        //                      lanes 16-31 -> K 8..15 & 24..31
        const int arow = mw + (lane & 15);
        const int ac0  = (lane >> 4) << 3;            // 0 or 8
        v16bf a0 = load_frag(&Als[(arow +  0) * LDSS + ac0],
                             &Als[(arow +  0) * LDSS + ac0 + 16]);
        v16bf a1 = load_frag(&Als[(arow + 16) * LDSS + ac0],
                             &Als[(arow + 16) * LDSS + ac0 + 16]);
        // B frag (32x16 bf16): lanes 0-15 -> N=lane, K 0..7 & 16..23; lanes 16-31 -> K 8..15 & 24..31
        const int bcol = nw + (lane & 15);
        const int bk0  = (lane >> 4) << 3;            // 0 or 8
        v16bf bb[4];
#pragma unroll
        for (int j = 0; j < 4; ++j)
            bb[j] = load_frag(&Bls[(bcol + j * 16) * LDSS + bk0],
                              &Bls[(bcol + j * 16) * LDSS + bk0 + 16]);

#pragma unroll
        for (int i = 0; i < 2; ++i) {
            v16bf ai = i ? a1 : a0;
#pragma unroll
            for (int j = 0; j < 4; ++j)
                acc[i][j] = __builtin_amdgcn_wmma_f32_16x16x32_bf16(
                    false, ai, false, bb[j], (short)0, acc[i][j], false, false);
        }

        if (t + 1 < T)     // convert+store next tile into the other buffer
            store_tiles(AlsBuf[(t + 1) & 1], BlsBuf[(t + 1) & 1], tid, ar, br);
    }

    // epilogue: C/D layout -> VGPR r: lanes 0-15 row=r, lanes 16-31 row=r+8
    const int nloc = lane & 15;
    const int rofs = (lane >> 4) << 3;
#pragma unroll
    for (int i = 0; i < 2; ++i) {
#pragma unroll
        for (int j = 0; j < 4; ++j) {
            const int ncol = n_blk + nw + j * 16 + nloc;
#pragma unroll
            for (int r = 0; r < 8; ++r) {
                const int mrow = m_blk + mw + i * 16 + rofs + r;
                float val = acc[i][j][r];
                if (R) val += R[((long)b * M + mrow) * 4096 + ncol];
                C[((long)b * cChan + chOff + mrow) * 4096 + ncol] = val;
            }
        }
    }
}

// depthwise 5x5, pad=2, on channels [0,1536) of qkvcat
__global__ __launch_bounds__(256) void dwconv5(
    const float* __restrict__ qkvcat, const float* __restrict__ wdw,
    float* __restrict__ out)
{
    const int hw = blockIdx.x * 256 + threadIdx.x;
    const int ch = blockIdx.y;
    const int b  = blockIdx.z;
    const int y = hw >> 6, x = hw & 63;
    const float* in = qkvcat + ((long)b * 3072 + ch) * 4096;
    const float* w  = wdw + ch * 25;
    float acc = 0.f;
#pragma unroll
    for (int dy = -2; dy <= 2; ++dy) {
        const int yy = y + dy;
        if (yy < 0 || yy > 63) continue;
#pragma unroll
        for (int dx = -2; dx <= 2; ++dx) {
            const int xx = x + dx;
            if (xx < 0 || xx > 63) continue;
            acc += w[(dy + 2) * 5 + (dx + 2)] * in[yy * 64 + xx];
        }
    }
    out[((long)b * 1536 + ch) * 4096 + hw] = acc;
}

// grouped 1x1: 192 groups of 8->8; writes channels [1536,3072) of qkvcat
__global__ __launch_bounds__(256) void pwgroup(
    const float* __restrict__ dw, const float* __restrict__ wpw,
    float* __restrict__ qkvcat)
{
    const int hw = blockIdx.x * 256 + threadIdx.x;
    const int co = blockIdx.y;
    const int b  = blockIdx.z;
    const int g  = co >> 3;
    const float* in = dw + ((long)b * 1536 + g * 8) * 4096 + hw;
    const float* w  = wpw + co * 8;
    float acc = 0.f;
#pragma unroll
    for (int ci = 0; ci < 8; ++ci) acc += w[ci] * in[(long)ci * 4096];
    qkvcat[((long)b * 3072 + 1536 + co) * 4096 + hw] = acc;
}

// per-(b, head) LiteMLA: vk = [v;1] k^T (9x8 over n=4096), out = (vk q) / (row8 + eps)
__global__ __launch_bounds__(256) void litemla_attn(
    const float* __restrict__ qkvcat, float* __restrict__ out)
{
    __shared__ float vk[72];           // [d 0..8][e 0..7]
    const int tid = threadIdx.x;
    const int h = blockIdx.x, b = blockIdx.y;
    if (tid < 72) vk[tid] = 0.f;
    __syncthreads();

    const float* base = qkvcat + ((long)b * 3072 + h * 24) * 4096;
    const float* Q = base;
    const float* Kp = base + 8 * 4096;
    const float* Vp = base + 16 * 4096;

    const int e = tid & 7;
    float a[9] = {};
    for (int n = tid >> 3; n < 4096; n += 32) {
        float kv = Kp[e * 4096 + n];
        kv = kv > 0.f ? kv : 0.f;
#pragma unroll
        for (int d = 0; d < 8; ++d) a[d] += Vp[d * 4096 + n] * kv;
        a[8] += kv;                     // v padded with 1.0 row
    }
#pragma unroll
    for (int d = 0; d < 9; ++d) atomicAdd(&vk[d * 8 + e], a[d]);
    __syncthreads();

    for (int n = tid; n < 4096; n += 256) {
        float q[8];
#pragma unroll
        for (int ee = 0; ee < 8; ++ee) {
            float v = Q[ee * 4096 + n];
            q[ee] = v > 0.f ? v : 0.f;
        }
        float den = 1e-15f;
#pragma unroll
        for (int ee = 0; ee < 8; ++ee) den += vk[64 + ee] * q[ee];
        const float inv = 1.0f / den;
#pragma unroll
        for (int d = 0; d < 8; ++d) {
            float o = 0.f;
#pragma unroll
            for (int ee = 0; ee < 8; ++ee) o += vk[d * 8 + ee] * q[ee];
            out[((long)b * 1024 + h * 8 + d) * 4096 + n] = o * inv;
        }
    }
}

extern "C" void kernel_launch(void* const* d_in, const int* in_sizes, int n_in,
                              void* d_out, int out_size, void* d_ws, size_t ws_size,
                              hipStream_t stream) {
    const float* x      = (const float*)d_in[0];   // [4,512,64,64]
    const float* w_qkv  = (const float*)d_in[1];   // [1536,512]
    const float* w_dw   = (const float*)d_in[2];   // [1536,25]
    const float* w_pw   = (const float*)d_in[3];   // [1536,8]
    const float* w_proj = (const float*)d_in[4];   // [512,1024]
    float* out = (float*)d_out;                    // [4,512,64,64]

    char* ws = (char*)d_ws;
    float* qkvcat  = (float*)ws;                                    // [4][3072][4096]
    float* dwout   = (float*)(ws + (size_t)4 * 3072 * 4096 * 4);    // [4][1536][4096]
    float* attnout = dwout;  // safe reuse: dwout fully consumed before attn writes

    // 1) qkv = x * w_qkv^T : M=1536 K=512, into channels [0,1536) of qkvcat
    gemm_wmma<<<dim3(4096 / TILE_N, 1536 / TILE_M, 4), 256, 0, stream>>>(
        w_qkv, x, qkvcat, nullptr, 1536, 512, 3072, 0);

    // 2) depthwise 5x5 on qkv
    dwconv5<<<dim3(16, 1536, 4), 256, 0, stream>>>(qkvcat, w_dw, dwout);

    // 3) grouped pointwise -> channels [1536,3072) of qkvcat (the concat)
    pwgroup<<<dim3(16, 1536, 4), 256, 0, stream>>>(dwout, w_pw, qkvcat);

    // 4) linear attention per (batch, head)
    litemla_attn<<<dim3(128, 4), 256, 0, stream>>>(qkvcat, attnout);

    // 5) out = x + attnout * w_proj^T : M=512 K=1024, residual add
    gemm_wmma<<<dim3(4096 / TILE_N, 512 / TILE_M, 4), 256, 0, stream>>>(
        w_proj, attnout, out, x, 512, 1024, 512, 0);
}
